// DiffusionReturnPrediction_78125455114518
// MI455X (gfx1250) — matrix operand order repaired and checked by reading, not compiled, not run on hardware
//
#include <hip/hip_runtime.h>

#define B_SZ    8
#define N_NODES 2000
#define T_STEPS 64
#define F_IN    32
#define HID     128
#define G4      512               // 4*HID
#define E_EDGES 32000
#define M_TOT   (B_SZ * N_NODES)  // 16000

typedef __attribute__((ext_vector_type(16))) __bf16 v16bf;
typedef __attribute__((ext_vector_type(8)))  float  v8f;
typedef __attribute__((ext_vector_type(2)))  __bf16 v2bf;
typedef __attribute__((ext_vector_type(2)))  float  v2f;

union Frag16 { v16bf v; unsigned u[8]; };

// single f32 -> bf16 (RNE) for the prep kernel
__device__ __forceinline__ unsigned short f2bf(float a) {
  __bf16 r = (__bf16)a;
  return *(unsigned short*)&r;
}
// packed pair f32x2 -> bf16x2; should lower to v_cvt_pk_bf16_f32
__device__ __forceinline__ unsigned pack_bf16x2(float a, float b) {
  v2f f; f[0] = a; f[1] = b;
  v2bf r = __builtin_convertvector(f, v2bf);
  return *(unsigned*)&r;
}
__device__ __forceinline__ float sigmoid_f(float x) { return 1.0f / (1.0f + __expf(-x)); }
__device__ __forceinline__ float tanh_f(float x)    { return 1.0f - 2.0f / (__expf(2.0f * x) + 1.0f); }

// B-matrix (16-bit, 32x16) kbase for VGPR v, lane-half hi  (table order, matches A)
__device__ __forceinline__ int bkb(int v, int hi) { return hi * 8 + (v & 3) * 2 + (v >> 2) * 16; }

// ---------------------------------------------------------------- prep -----
__global__ __launch_bounds__(256) void k_prep(
    const float* __restrict__ Wih, const float* __restrict__ Whh,
    const float* __restrict__ gW,  const float* __restrict__ bih,
    const float* __restrict__ bhh,
    unsigned short* __restrict__ WihB, unsigned short* __restrict__ WhhB,
    unsigned short* __restrict__ WtB,  float* __restrict__ bc) {
  int i = blockIdx.x * 256 + threadIdx.x;
  if (i < 16384) {                          // Wih [512][32] -> bf16
    WihB[i] = f2bf(Wih[i]);
  } else if (i < 81920) {                   // Whh [512][128] -> bf16
    int j = i - 16384; WhhB[j] = f2bf(Whh[j]);
  } else if (i < 98304) {                   // gcn_W [128][128] -> bf16 transposed: Wt[n][k]
    int j = i - 81920; int n = j >> 7, k = j & 127;
    WtB[n * 128 + k] = f2bf(gW[k * 128 + n]);
  } else if (i < 98816) {                   // combined LSTM bias
    int j = i - 98304; bc[j] = bih[j] + bhh[j];
  }
}

// ---------------------------------------------------------------- LSTM -----
// One 256-thread block (8 waves) per 16-row M tile. Wave w owns gate
// columns [64w, 64w+64): 4 N-tiles. All B fragments VGPR-resident.
// x_t and h live in LDS as bf16 so A-fragment build is pure ds_load_b128.
__global__ __launch_bounds__(256) void k_lstm(
    const float* __restrict__ x,              // [M][T][F] f32
    const unsigned short* __restrict__ WihB,  // [512][32] bf16 row-major
    const unsigned short* __restrict__ WhhB,  // [512][128] bf16 row-major
    const float* __restrict__ bc,             // [512]
    unsigned short* __restrict__ hout) {      // [M][128] bf16
  __shared__ __align__(16) unsigned short xbufB[16 * F_IN];  // x_t tile, bf16
  __shared__ __align__(16) unsigned short hbufB[16 * HID];   // h, bf16
  __shared__ __align__(16) float zbuf[G4 * 16];              // z, [n][m]

  const int tid  = threadIdx.x;
  const int lane = tid & 31;
  const int wave = tid >> 5;        // 0..7
  const int m0   = blockIdx.x * 16;
  const int ln   = lane & 15;
  const int hi   = lane >> 4;
  const int cb   = wave * 64;

  // ---- preload resident B fragments (Wih^T and Whh^T as 32x16 bf16 tiles)
  Frag16 Bx[4];
  Frag16 Bh[4][4];                   // [ntile][ktile]
  float  bias[4];
#pragma unroll
  for (int nt = 0; nt < 4; ++nt) {
    const int n = cb + nt * 16 + ln;
    const unsigned short* wr = WihB + n * F_IN;
    const unsigned short* hr = WhhB + n * HID;
#pragma unroll
    for (int v = 0; v < 8; ++v)
      Bx[nt].u[v] = *(const unsigned*)(wr + bkb(v, hi));
#pragma unroll
    for (int kt = 0; kt < 4; ++kt)
#pragma unroll
      for (int v = 0; v < 8; ++v)
        Bh[nt][kt].u[v] = *(const unsigned*)(hr + kt * 32 + bkb(v, hi));
    bias[nt] = bc[n];
  }

  // ---- init state (h = 0 as bf16, c = 0 in registers)
  for (int i = tid; i < 16 * HID / 2; i += 256) ((unsigned*)hbufB)[i] = 0u;
  float creg[8];
#pragma unroll
  for (int e = 0; e < 8; ++e) creg[e] = 0.0f;

  const int gm = tid & 15;           // gate-phase row
  const int jb = (tid >> 4) * 8;     // gate-phase hidden-col base

  for (int t = 0; t < T_STEPS; ++t) {
    // stage x_t tile [16][32] -> LDS bf16 (2 floats -> 1 packed dword / thread)
    {
      int idx = tid * 2;
      int r = idx >> 5, c = idx & 31;
      const float* xp = x + ((size_t)(m0 + r) * T_STEPS + t) * F_IN + c;
      float2 xv = *(const float2*)xp;
      *(unsigned*)&xbufB[idx] = pack_bf16x2(xv.x, xv.y);
    }
    __syncthreads();   // xbufB ready; hbufB from previous step ready

    // build A fragments: pure 16-byte LDS loads (layout ISA 7.12.2:
    // VGPR v=0..3 hold K=hi*8+{0..7}; v=4..7 hold K=16+hi*8+{0..7})
    Frag16 Ax;
    Frag16 Ah[4];
    {
      const unsigned short* xr = &xbufB[ln * F_IN + hi * 8];
      *(uint4*)&Ax.u[0] = *(const uint4*)xr;
      *(uint4*)&Ax.u[4] = *(const uint4*)(xr + 16);
    }
#pragma unroll
    for (int kt = 0; kt < 4; ++kt) {
      const unsigned short* hr = &hbufB[ln * HID + kt * 32 + hi * 8];
      *(uint4*)&Ah[kt].u[0] = *(const uint4*)hr;
      *(uint4*)&Ah[kt].u[4] = *(const uint4*)(hr + 16);
    }

    // z = bias + x_t@Wih^T + h@Whh^T  (20 WMMA / wave / step)
#pragma unroll
    for (int nt = 0; nt < 4; ++nt) {
      v8f acc;
#pragma unroll
      for (int e = 0; e < 8; ++e) acc[e] = bias[nt];
      acc = __builtin_amdgcn_wmma_f32_16x16x32_bf16(false, Ax.v, false, Bx[nt].v,
                                                    (short)0, acc, false, false);
#pragma unroll
      for (int kt = 0; kt < 4; ++kt)
        acc = __builtin_amdgcn_wmma_f32_16x16x32_bf16(false, Ah[kt].v, false, Bh[nt][kt].v,
                                                      (short)0, acc, false, false);
      // D layout: lane ln = n, rows m = v + 8*hi  -> zbuf[n][m]
      const int n = cb + nt * 16 + ln;
      float* zp = &zbuf[n * 16 + hi * 8];
      *(float4*)zp       = make_float4(acc[0], acc[1], acc[2], acc[3]);
      *(float4*)(zp + 4) = make_float4(acc[4], acc[5], acc[6], acc[7]);
    }
    __syncthreads();   // zbuf ready; hbufB reads complete

    // elementwise gates: thread owns (m=gm, j=jb..jb+7); c in registers
    float hv[8];
#pragma unroll
    for (int e = 0; e < 8; ++e) {
      const int j = jb + e;
      const float iz = zbuf[(j      ) * 16 + gm];
      const float fz = zbuf[(j + 128) * 16 + gm];
      const float gz = zbuf[(j + 256) * 16 + gm];
      const float oz = zbuf[(j + 384) * 16 + gm];
      const float c  = sigmoid_f(fz) * creg[e] + sigmoid_f(iz) * tanh_f(gz);
      creg[e] = c;
      hv[e]   = sigmoid_f(oz) * tanh_f(c);
    }
    // write h back as bf16 (4 packed dwords -> one 16B LDS store)
    {
      uint4 hq;
      hq.x = pack_bf16x2(hv[0], hv[1]);
      hq.y = pack_bf16x2(hv[2], hv[3]);
      hq.z = pack_bf16x2(hv[4], hv[5]);
      hq.w = pack_bf16x2(hv[6], hv[7]);
      *(uint4*)&hbufB[gm * HID + jb] = hq;
    }
    // next iteration's first barrier orders hbufB writes vs. A-fragment reads
  }

  __syncthreads();
  // write final h tile (contiguous 2048 bf16 = 4KB)
  *(uint4*)&hout[(size_t)m0 * HID + tid * 8] = *(const uint4*)&hbufB[tid * 8];
}

// ----------------------------------------------------------- GCN: degree ---
__global__ __launch_bounds__(256) void k_deg_init(float* __restrict__ deg) {
  int n = blockIdx.x * 256 + threadIdx.x;
  if (n < N_NODES) deg[n] = 1.0f;      // self-loop
}
__global__ __launch_bounds__(256) void k_deg_edges(const int* __restrict__ ei,
                                                   float* __restrict__ deg) {
  int e = blockIdx.x * 256 + threadIdx.x;
  if (e < E_EDGES) atomicAdd(&deg[ei[E_EDGES + e]], 1.0f);
}
__global__ __launch_bounds__(256) void k_dinv(const float* __restrict__ deg,
                                              float* __restrict__ dinv) {
  int n = blockIdx.x * 256 + threadIdx.x;
  if (n < N_NODES) dinv[n] = rsqrtf(deg[n]);
}

// ------------------------------------------------- GCN: xw = h @ gcn_W -----
__global__ __launch_bounds__(128) void k_xw(
    const unsigned short* __restrict__ hin,   // [M][128] bf16
    const unsigned short* __restrict__ WtB,   // [128][128] bf16, Wt[n][k]
    float* __restrict__ xw) {                 // [M][128] f32
  __shared__ __align__(16) unsigned short hb[16 * HID];
  const int tid = threadIdx.x, lane = tid & 31, wave = tid >> 5;
  const int m0 = blockIdx.x * 16;
  const int ln = lane & 15, hi = lane >> 4;
  for (int i = tid * 8; i < 16 * HID; i += 128 * 8)
    *(uint4*)&hb[i] = *(const uint4*)&hin[(size_t)m0 * HID + i];
  __syncthreads();
  const int cb = wave * 32;
#pragma unroll
  for (int nt = 0; nt < 2; ++nt) {
    const int n = cb + nt * 16 + ln;
    const unsigned short* wr = WtB + n * HID;
    v8f acc = {};
#pragma unroll
    for (int kt = 0; kt < 4; ++kt) {
      Frag16 A, Bf;
      const unsigned short* hr = &hb[ln * HID + kt * 32 + hi * 8];
      *(uint4*)&A.u[0] = *(const uint4*)hr;
      *(uint4*)&A.u[4] = *(const uint4*)(hr + 16);
#pragma unroll
      for (int v = 0; v < 8; ++v)
        Bf.u[v] = *(const unsigned*)(wr + kt * 32 + bkb(v, hi));
      acc = __builtin_amdgcn_wmma_f32_16x16x32_bf16(false, A.v, false, Bf.v,
                                                    (short)0, acc, false, false);
    }
#pragma unroll
    for (int v = 0; v < 8; ++v)
      xw[(size_t)(m0 + v + hi * 8) * HID + n] = acc[v];
  }
}

// ----------------------------------------------------- GCN: aggregation ----
__global__ __launch_bounds__(128) void k_agg_init(const float* __restrict__ xw,
                                                  const float* __restrict__ dinv,
                                                  float* __restrict__ agg) {
  const int node = blockIdx.x;          // b*2000+n
  const int n = node % N_NODES;
  const float w = dinv[n] * dinv[n];
  const int j = threadIdx.x;
  agg[(size_t)node * HID + j] = xw[(size_t)node * HID + j] * w;
}
__global__ __launch_bounds__(128) void k_agg_edges(const int* __restrict__ ei,
                                                   const float* __restrict__ dinv,
                                                   const float* __restrict__ xw,
                                                   float* __restrict__ agg) {
  const int e = blockIdx.x;
  const int s = ei[e];
  const int d = ei[E_EDGES + e];
  const float w = dinv[s] * dinv[d];
  const int j = threadIdx.x;
#pragma unroll
  for (int b = 0; b < B_SZ; ++b) {
    const float m = xw[((size_t)b * N_NODES + s) * HID + j] * w;
    atomicAdd(&agg[((size_t)b * N_NODES + d) * HID + j], m);
  }
}

// ------------------------------------------------------------- MLP head ----
__global__ __launch_bounds__(64) void k_mlp(const float* __restrict__ agg,
                                            const float* __restrict__ gcn_b,
                                            const float* __restrict__ W1,
                                            const float* __restrict__ b1,
                                            const float* __restrict__ W2,
                                            const float* __restrict__ b2,
                                            float* __restrict__ out) {
  __shared__ __align__(8) float arow[HID];
  __shared__ float psum[2];
  const int tid  = threadIdx.x;
  const int node = blockIdx.x;
  {
    float2 v = *(const float2*)(agg + (size_t)node * HID + tid * 2);
    float2 g = *(const float2*)(gcn_b + tid * 2);
    arow[tid * 2]     = v.x + g.x;
    arow[tid * 2 + 1] = v.y + g.y;
  }
  __syncthreads();
  float acc = b1[tid];
#pragma unroll 8
  for (int k = 0; k < HID; ++k) acc = fmaf(arow[k], W1[k * 64 + tid], acc);
  float val = (acc * sigmoid_f(acc)) * W2[tid];   // silu(acc) * W2
#pragma unroll
  for (int off = 16; off > 0; off >>= 1) val += __shfl_xor(val, off, 32);
  if ((tid & 31) == 0) psum[tid >> 5] = val;
  __syncthreads();
  if (tid == 0) out[node] = psum[0] + psum[1] + b2[0];
}

// --------------------------------------------------------------- launch ----
extern "C" void kernel_launch(void* const* d_in, const int* in_sizes, int n_in,
                              void* d_out, int out_size, void* d_ws, size_t ws_size,
                              hipStream_t stream) {
  (void)in_sizes; (void)n_in; (void)out_size; (void)ws_size;
  const float* x   = (const float*)d_in[0];
  const int*   ei  = (const int*)d_in[1];
  const float* Wih = (const float*)d_in[2];
  const float* Whh = (const float*)d_in[3];
  const float* bih = (const float*)d_in[4];
  const float* bhh = (const float*)d_in[5];
  const float* gW  = (const float*)d_in[6];
  const float* gb  = (const float*)d_in[7];
  const float* W1  = (const float*)d_in[8];
  const float* b1  = (const float*)d_in[9];
  const float* W2  = (const float*)d_in[10];
  const float* b2  = (const float*)d_in[11];
  float* out = (float*)d_out;

  char* wsp = (char*)d_ws;
  auto take = [&](size_t bytes) -> char* {
    char* p = wsp;
    wsp += (bytes + 255) & ~(size_t)255;
    return p;
  };
  unsigned short* WihB = (unsigned short*)take((size_t)G4 * F_IN * 2);
  unsigned short* WhhB = (unsigned short*)take((size_t)G4 * HID * 2);
  unsigned short* WtB  = (unsigned short*)take((size_t)HID * HID * 2);
  float*          bc   = (float*)take((size_t)G4 * 4);
  unsigned short* hout = (unsigned short*)take((size_t)M_TOT * HID * 2);
  float*          xw   = (float*)take((size_t)M_TOT * HID * 4);
  float*          agg  = (float*)take((size_t)M_TOT * HID * 4);
  float*          deg  = (float*)take((size_t)N_NODES * 4);
  float*          dinv = (float*)take((size_t)N_NODES * 4);

  k_prep<<<386, 256, 0, stream>>>(Wih, Whh, gW, bih, bhh, WihB, WhhB, WtB, bc);
  k_lstm<<<M_TOT / 16, 256, 0, stream>>>(x, WihB, WhhB, bc, hout);
  k_deg_init<<<(N_NODES + 255) / 256, 256, 0, stream>>>(deg);
  k_deg_edges<<<(E_EDGES + 255) / 256, 256, 0, stream>>>(ei, deg);
  k_dinv<<<(N_NODES + 255) / 256, 256, 0, stream>>>(deg, dinv);
  k_xw<<<M_TOT / 16, 128, 0, stream>>>(hout, WtB, xw);
  k_agg_init<<<M_TOT, 128, 0, stream>>>(xw, dinv, agg);
  k_agg_edges<<<E_EDGES, 128, 0, stream>>>(ei, dinv, xw, agg);
  k_mlp<<<M_TOT, 64, 0, stream>>>(agg, gb, W1, b1, W2, b2, out);
}
